// message_passing_gnn__77197742178597
// MI455X (gfx1250) — matrix operand order repaired and checked by reading, not compiled, not run on hardware
//
#include <hip/hip_runtime.h>
#include <hip/hip_bf16.h>

typedef __bf16 bf16;
typedef __attribute__((ext_vector_type(16))) __bf16 v16bf;
typedef __attribute__((ext_vector_type(8)))  __bf16 v8bf;
typedef __attribute__((ext_vector_type(8)))  float  v8f;

#define NODES 50000
#define EDGES 800000
#define DIN   256
#define DD    128
#define NB    8

// ---------------------------------------------------------------------------
// WMMA helpers (V_WMMA_F32_16X16X32_BF16, wave32)
// A layout (16x32 bf16): lane<16 -> row=lane, K kb+0..15
//                        lane>=16 -> row=lane-16, K kb+16..31
// B layout (32x16 bf16): col = lane&15; lanes 0-15 hold K kb+0..15,
//                        lanes 16-31 hold K kb+16..31 (contiguous per lane
//                        because weights are pre-packed [N_out][K]).
// C/D layout: col = lane&15; VGPR r -> row r (lanes 0-15) / r+8 (lanes 16-31)
// ---------------------------------------------------------------------------
__device__ __forceinline__ v8f zero8() {
  v8f z;
#pragma unroll
  for (int i = 0; i < 8; ++i) z[i] = 0.0f;
  return z;
}

__device__ __forceinline__ v16bf load_a(const bf16* rowp, int kb, int hi) {
  v16bf a;
  const int ko = kb + hi * 16;
  ((v8bf*)&a)[0] = *(const v8bf*)(rowp + ko);
  ((v8bf*)&a)[1] = *(const v8bf*)(rowp + ko + 8);
  return a;
}

__device__ __forceinline__ v16bf load_b(const bf16* colp, int kb, int hi) {
  v16bf b;
  const v8bf* p = (const v8bf*)(colp + kb + hi * 16);
  ((v8bf*)&b)[0] = p[0];
  ((v8bf*)&b)[1] = p[1];
  return b;
}

__device__ __forceinline__ v8f wmma_bf16(v16bf a, v16bf b, v8f c) {
  return __builtin_amdgcn_wmma_f32_16x16x32_bf16(false, a, false, b, (short)0, c,
                                                 false, false);
}

// ---------------------------------------------------------------------------
// Generic wave-level GEMM: C[M,NOUT] = act(A_bf16[M,K] @ Bp[NOUT,K]^T + bias)
// One wave computes a 16 x NOUT stripe. Optional residual, f32 and bf16 outs.
// ---------------------------------------------------------------------------
template <int K, int NOUT, bool RELU>
__global__ __launch_bounds__(32) void k_gemm(const bf16* __restrict__ A,
                                             const bf16* __restrict__ Bp,
                                             const float* __restrict__ bias,
                                             float* __restrict__ C,
                                             bf16* __restrict__ Cb,
                                             const float* __restrict__ resid) {
  constexpr int NT = NOUT / 16;
  constexpr int KS = K / 32;
  const int lane = threadIdx.x;
  const int row  = lane & 15;
  const int hi   = lane >> 4;
  const size_t m0 = (size_t)blockIdx.x * 16;
  const bf16* arow = A + (m0 + row) * (size_t)K;

  v8f acc[NT];
#pragma unroll
  for (int t = 0; t < NT; ++t) acc[t] = zero8();

#pragma unroll
  for (int ks = 0; ks < KS; ++ks) {
    const v16bf a = load_a(arow, ks * 32, hi);
#pragma unroll
    for (int t = 0; t < NT; ++t) {
      const v16bf b = load_b(Bp + (size_t)(t * 16 + row) * K, ks * 32, hi);
      acc[t] = wmma_bf16(a, b, acc[t]);
    }
  }

#pragma unroll
  for (int t = 0; t < NT; ++t) {
    const int n = t * 16 + row;
    const float bb = bias[n];
#pragma unroll
    for (int r = 0; r < 8; ++r) {
      const size_t m = m0 + (size_t)(r + hi * 8);
      float v = acc[t][r] + bb;
      if (RELU) v = v > 0.0f ? v : 0.0f;
      if (resid) v += resid[m * NOUT + n];
      if (C)  C[m * NOUT + n] = v;
      if (Cb) Cb[m * NOUT + n] = (bf16)v;
    }
  }
}

// ---------------------------------------------------------------------------
// Fused edge MLP, 32 edges (two 16-row A-tiles) per wave so that every
// weight (B) fragment load feeds TWO WMMAs:
//   h = cat(x[idx_i], x[idx_j])            gathered, K=256
//   m = relu(h @ W1 + b1) @ W2 + b2        WMMA, hidden via LDS transpose
//   agg[idx_i] += m * inv_deg[idx_i]       fused f32 atomic scatter
// ---------------------------------------------------------------------------
__global__ __launch_bounds__(32) void k_edge_mlp(const bf16* __restrict__ x,
                                                 const int* __restrict__ idx_i,
                                                 const int* __restrict__ idx_j,
                                                 const bf16* __restrict__ W1p,
                                                 const float* __restrict__ b1,
                                                 const bf16* __restrict__ W2p,
                                                 const float* __restrict__ b2,
                                                 const float* __restrict__ invdeg,
                                                 float* __restrict__ agg) {
  __shared__ bf16 lh[32 * DD];  // 8 KB: hidden activations for 32 edges
  const int lane = threadIdx.x;
  const int row  = lane & 15;
  const int hi   = lane >> 4;
  const int e0   = blockIdx.x * 32;

  const bf16* rowi[2];
  const bf16* rowj[2];
#pragma unroll
  for (int u = 0; u < 2; ++u) {
    rowi[u] = x + (size_t)idx_i[e0 + u * 16 + row] * DD;
    rowj[u] = x + (size_t)idx_j[e0 + u * 16 + row] * DD;
  }

  v8f acc[2][8];
#pragma unroll
  for (int u = 0; u < 2; ++u)
#pragma unroll
    for (int t = 0; t < 8; ++t) acc[u][t] = zero8();

  // GEMM1: K = 256; steps 0-3 gather from x[idx_i], steps 4-7 from x[idx_j]
#pragma unroll
  for (int ks = 0; ks < 8; ++ks) {
    v16bf a[2];
#pragma unroll
    for (int u = 0; u < 2; ++u)
      a[u] = load_a((ks < 4) ? rowi[u] : rowj[u], (ks & 3) * 32, hi);
#pragma unroll
    for (int t = 0; t < 8; ++t) {
      const v16bf b = load_b(W1p + (size_t)(t * 16 + row) * 256, ks * 32, hi);
      acc[0][t] = wmma_bf16(a[0], b, acc[0][t]);
      acc[1][t] = wmma_bf16(a[1], b, acc[1][t]);
    }
  }

  // bias + ReLU + C-layout -> row-major bf16 in LDS (A-layout source for GEMM2)
#pragma unroll
  for (int t = 0; t < 8; ++t) {
    const int n = t * 16 + row;
    const float bb = b1[n];
#pragma unroll
    for (int u = 0; u < 2; ++u) {
#pragma unroll
      for (int r = 0; r < 8; ++r) {
        float v = acc[u][t][r] + bb;
        v = v > 0.0f ? v : 0.0f;
        lh[(u * 16 + r + hi * 8) * DD + n] = (bf16)v;
      }
    }
  }
  __syncthreads();

  // GEMM2: K = 128 from LDS
  v8f acc2[2][8];
#pragma unroll
  for (int u = 0; u < 2; ++u)
#pragma unroll
    for (int t = 0; t < 8; ++t) acc2[u][t] = zero8();
#pragma unroll
  for (int ks = 0; ks < 4; ++ks) {
    v16bf a[2];
#pragma unroll
    for (int u = 0; u < 2; ++u)
      a[u] = load_a(&lh[(u * 16 + row) * DD], ks * 32, hi);
#pragma unroll
    for (int t = 0; t < 8; ++t) {
      const v16bf b = load_b(W2p + (size_t)(t * 16 + row) * DD, ks * 32, hi);
      acc2[0][t] = wmma_bf16(a[0], b, acc2[0][t]);
      acc2[1][t] = wmma_bf16(a[1], b, acc2[1][t]);
    }
  }

  // (m + b2) * inv_deg[i], atomic scatter-add (== segment_sum then normalize)
#pragma unroll
  for (int u = 0; u < 2; ++u) {
#pragma unroll
    for (int r = 0; r < 8; ++r) {
      const int e  = e0 + u * 16 + r + hi * 8;
      const int ti = idx_i[e];
      const float sc = invdeg[ti];
      float* dst = agg + (size_t)ti * DD;
#pragma unroll
      for (int t = 0; t < 8; ++t) {
        const int n = t * 16 + row;
        atomicAdd(dst + n, (acc2[u][t][r] + b2[n]) * sc);
      }
    }
  }
}

// ------------------------- small utility kernels ---------------------------
__global__ void k_pack(const float* __restrict__ W, bf16* __restrict__ Wp,
                       int K, int Nout) {  // Wp[n][k] = W[k][n]
  int i = blockIdx.x * 256 + threadIdx.x;
  if (i >= K * Nout) return;
  int n = i / K, k = i - n * K;
  Wp[i] = (bf16)W[(size_t)k * Nout + n];
}

__global__ void k_cvt(const float* __restrict__ in, bf16* __restrict__ out, int n) {
  int i = blockIdx.x * 256 + threadIdx.x;
  if (i < n) out[i] = (bf16)in[i];
}

__global__ void k_deg(const int* __restrict__ idx, float* __restrict__ deg, int n) {
  int i = blockIdx.x * 256 + threadIdx.x;
  if (i < n) atomicAdd(&deg[idx[i]], 1.0f);
}

__global__ void k_inv(float* __restrict__ d, int n) {
  int i = blockIdx.x * 256 + threadIdx.x;
  if (i < n) d[i] = d[i] > 0.0f ? 1.0f / d[i] : 0.0f;
}

__global__ void k_hcat(const bf16* __restrict__ xb, const float* __restrict__ fi,
                       const float* __restrict__ fo, bf16* __restrict__ h, int n) {
  int i = blockIdx.x * 256 + threadIdx.x;
  if (i >= n) return;  // n = NODES*DD
  int node = i / DD, c = i - node * DD;
  bf16* hr = h + (size_t)node * (3 * DD);
  hr[c] = xb[i];
  hr[DD + c] = (bf16)fi[i];
  hr[2 * DD + c] = (bf16)fo[i];
}

__global__ void k_bounds(const int* __restrict__ batch, int* __restrict__ st,
                         int* __restrict__ en, int n) {  // batch is sorted
  int i = blockIdx.x * 256 + threadIdx.x;
  if (i >= n) return;
  if (i == 0) st[batch[0]] = 0;
  else if (batch[i] != batch[i - 1]) { st[batch[i]] = i; en[batch[i - 1]] = i; }
  if (i == n - 1) en[batch[i]] = n;
}

__global__ void k_pool(const float* __restrict__ y, const int* __restrict__ st,
                       const int* __restrict__ en, float* __restrict__ out) {
  int b = blockIdx.x, c = threadIdx.x;  // 256 threads = 2*DD channels
  int s = st[b], e = en[b];
  float mx = -__builtin_inff(), sm = 0.0f;
  for (int n = s; n < e; ++n) {
    float v = y[(size_t)n * (2 * DD) + c];
    mx = fmaxf(mx, v);
    sm += v;
  }
  out[(size_t)b * (4 * DD) + c] = mx;
  out[(size_t)b * (4 * DD) + 2 * DD + c] = sm / (float)(e - s);
}

// ---------------------------------------------------------------------------
extern "C" void kernel_launch(void* const* d_in, const int* in_sizes, int n_in,
                              void* d_out, int out_size, void* d_ws, size_t ws_size,
                              hipStream_t stream) {
  const float* nodes = (const float*)d_in[0];
  const int*   edges = (const int*)d_in[1];
  const int*   batch = (const int*)d_in[2];
  const float* W_enc = (const float*)d_in[3];
  const float* b_enc = (const float*)d_in[4];
  const float* Wp1 = (const float*)d_in[5];  const float* bp1 = (const float*)d_in[6];
  const float* Wp2 = (const float*)d_in[7];  const float* bp2 = (const float*)d_in[8];
  const float* Wc1 = (const float*)d_in[9];  const float* bc1 = (const float*)d_in[10];
  const float* Wc2 = (const float*)d_in[11]; const float* bc2 = (const float*)d_in[12];
  const float* Wf1 = (const float*)d_in[13]; const float* bf1 = (const float*)d_in[14];
  const float* Wf2 = (const float*)d_in[15]; const float* bf2 = (const float*)d_in[16];
  const float* Wconv = (const float*)d_in[17]; const float* bconv = (const float*)d_in[18];
  float* out = (float*)d_out;
  (void)in_sizes; (void)n_in; (void)out_size; (void)ws_size;

  const int* e_src = edges;          // edges[0]
  const int* e_dst = edges + EDGES;  // edges[1]

  // ---- workspace carve-out (~155 MB, 256B-aligned blocks) ----
  char* ws = (char*)d_ws;
  size_t off = 0;
  auto alloc = [&](size_t bytes) {
    void* p = ws + off;
    off += (bytes + 255) & ~(size_t)255;
    return p;
  };
  float* inv_dst = (float*)alloc((size_t)NODES * 4);
  float* inv_src = (float*)alloc((size_t)NODES * 4);
  bf16*  nodes_b = (bf16*)alloc((size_t)NODES * DIN * 2);  // encoder input
  bf16*  t_b     = nodes_b;  // [N,256] hidden of final MLP; lifetime disjoint
  float* x_f     = (float*)alloc((size_t)NODES * DD * 4);
  bf16*  x_b     = (bf16*)alloc((size_t)NODES * DD * 2);
  float* fi      = (float*)alloc((size_t)NODES * DD * 4);
  float* fo      = (float*)alloc((size_t)NODES * DD * 4);  // contiguous after fi
  float* y       = fi;  // [N,256] conv out aliases fi+fo (both dead by then)
  bf16*  hcat    = (bf16*)alloc((size_t)NODES * 3 * DD * 2);
  int*   st      = (int*)alloc(NB * 4);
  int*   en      = (int*)alloc(NB * 4);
  bf16* pWenc  = (bf16*)alloc(256 * 128 * 2);
  bf16* pWp1   = (bf16*)alloc(256 * 128 * 2);
  bf16* pWp2   = (bf16*)alloc(128 * 128 * 2);
  bf16* pWc1   = (bf16*)alloc(256 * 128 * 2);
  bf16* pWc2   = (bf16*)alloc(128 * 128 * 2);
  bf16* pWf1   = (bf16*)alloc(384 * 256 * 2);
  bf16* pWf2   = (bf16*)alloc(256 * 128 * 2);
  bf16* pWconv = (bf16*)alloc(128 * 256 * 2);

  // ---- degrees -> inverse degrees ----
  hipMemsetAsync(inv_dst, 0, (size_t)NODES * 4, stream);
  hipMemsetAsync(inv_src, 0, (size_t)NODES * 4, stream);
  k_deg<<<EDGES / 256, 256, 0, stream>>>(e_dst, inv_dst, EDGES);
  k_deg<<<EDGES / 256, 256, 0, stream>>>(e_src, inv_src, EDGES);
  k_inv<<<(NODES + 255) / 256, 256, 0, stream>>>(inv_dst, NODES);
  k_inv<<<(NODES + 255) / 256, 256, 0, stream>>>(inv_src, NODES);

  // ---- weight pre-pack (transpose to [N_out][K], bf16) ----
  auto pack = [&](const float* W, bf16* P, int K, int Nout) {
    k_pack<<<(K * Nout + 255) / 256, 256, 0, stream>>>(W, P, K, Nout);
  };
  pack(W_enc, pWenc, 256, 128);
  pack(Wp1, pWp1, 256, 128);  pack(Wp2, pWp2, 128, 128);
  pack(Wc1, pWc1, 256, 128);  pack(Wc2, pWc2, 128, 128);
  pack(Wf1, pWf1, 384, 256);  pack(Wf2, pWf2, 256, 128);
  pack(Wconv, pWconv, 128, 256);

  // ---- encoder: x = nodes @ W_enc + b_enc ----
  k_cvt<<<(NODES * DIN + 255) / 256, 256, 0, stream>>>(nodes, nodes_b, NODES * DIN);
  k_gemm<256, 128, false><<<NODES / 16, 32, 0, stream>>>(nodes_b, pWenc, b_enc,
                                                         x_f, x_b, nullptr);

  // ---- message-passing iterations ----
  for (int it = 0; it < 2; ++it) {
    hipMemsetAsync(fi, 0, (size_t)NODES * DD * 4, stream);
    hipMemsetAsync(fo, 0, (size_t)NODES * DD * 4, stream);
    // parent: i=dst, j=src, normalize by in-degree
    k_edge_mlp<<<EDGES / 32, 32, 0, stream>>>(x_b, e_dst, e_src, pWp1, bp1,
                                              pWp2, bp2, inv_dst, fi);
    // child: i=src, j=dst, normalize by out-degree
    k_edge_mlp<<<EDGES / 32, 32, 0, stream>>>(x_b, e_src, e_dst, pWc1, bc1,
                                              pWc2, bc2, inv_src, fo);
    // h = cat(x, fi, fo);  x = x + relu(h@Wf1+bf1)@Wf2+bf2
    k_hcat<<<(NODES * DD + 255) / 256, 256, 0, stream>>>(x_b, fi, fo, hcat,
                                                         NODES * DD);
    k_gemm<384, 256, true><<<NODES / 16, 32, 0, stream>>>(hcat, pWf1, bf1,
                                                          nullptr, t_b, nullptr);
    k_gemm<256, 128, false><<<NODES / 16, 32, 0, stream>>>(t_b, pWf2, bf2,
                                                           x_f, x_b, x_f);
  }

  // ---- conv1d(k=1) + pooling ----
  k_gemm<128, 256, false><<<NODES / 16, 32, 0, stream>>>(x_b, pWconv, bconv,
                                                         y, nullptr, nullptr);
  hipMemsetAsync(st, 0, NB * 4, stream);
  hipMemsetAsync(en, 0, NB * 4, stream);
  k_bounds<<<(NODES + 255) / 256, 256, 0, stream>>>(batch, st, en, NODES);
  k_pool<<<NB, 256, 0, stream>>>(y, st, en, out);
}
